// MUCRP_17265768530653
// MI455X (gfx1250) — compile-verified, matrix operand order checked by reading, not run on hardware
//
#include <hip/hip_runtime.h>

// MI455X / gfx1250, wave32. WMMA bf16 GEMM pipeline for the dual-VAE + FGW forward.

typedef __bf16 bf16;
typedef __attribute__((ext_vector_type(16))) __bf16 v16bf;
typedef __attribute__((ext_vector_type(8)))  __bf16 v8bf;
typedef __attribute__((ext_vector_type(8)))  float  v8f;

#define LOG2PI_F 1.8378770664093454f

// ---------------------------------------------------------------------------
// Generic WMMA GEMM:  out = act( alpha * A[M,K] @ Bt[N,K]^T + bias [+ prev] )
// AF32: A operand is f32 in memory, converted to bf16 while staging (template
// constant -> no per-iteration branch). Tile 128x64x32, 256 threads = 8 waves,
// each wave owns a 32x32 patch (4 x v_wmma_f32_16x16x32_bf16 with A/B reuse).
// Software pipeline: step s+1's global tiles are prefetched into registers
// while step s's WMMAs execute; staging rows are clamped so the main loop is
// guard-free (over-reads land in rows/cols that are never stored); the K
// remainder is one guarded pipelined step.
// ---------------------------------------------------------------------------
template<int AF32>
__global__ void __launch_bounds__(256)
k_gemm_t(const void* __restrict__ A,
         const bf16* __restrict__ Bt,
         const float* __restrict__ bias,
         float* __restrict__ outF,
         bf16* __restrict__ outBf,
         float alpha, int relu, int accum,
         int M, int N, int Kd)
{
    __shared__ bf16 As[128][32];
    __shared__ bf16 Bs[64][32];

    const int tid  = threadIdx.x;
    const int wave = tid >> 5;
    const int lane = tid & 31;
    const int wr   = wave & 3;        // 4 row groups of 32
    const int wc   = wave >> 2;       // 2 col groups of 32
    const int m0   = blockIdx.y * 128;
    const int n0   = blockIdx.x * 64;
    const int r    = lane & 15;
    const int half = lane >> 4;

    const int lr = tid >> 2;          // 0..63: staging row
    const int lk = (tid & 3) * 8;     // k offset (8 elements per thread)

    // Clamped global rows: beyond-edge rows stage garbage that is never stored.
    const int gmA0 = min(m0 + lr,      M - 1);
    const int gmA1 = min(m0 + lr + 64, M - 1);
    const int gnB  = min(n0 + lr,      N - 1);

    v8f acc00 = {}, acc01 = {}, acc10 = {}, acc11 = {};
    v8bf ra0 = {}, ra1 = {}, rb = {};      // register staging (prefetch) buffers

    const int nfull  = Kd >> 5;
    const int nsteps = nfull + ((Kd & 31) ? 1 : 0);

    // stage step -> registers (fast path: unguarded vector loads)
    auto stage_fast = [&](int step) {
        const int gk = (step << 5) + lk;
        if (AF32) {
            const float* Af = (const float*)A;
            {
                const float4* p = (const float4*)(Af + (size_t)gmA0 * Kd + gk);
                float4 f0 = p[0], f1 = p[1];
                v8bf t;
                t[0]=(bf16)f0.x; t[1]=(bf16)f0.y; t[2]=(bf16)f0.z; t[3]=(bf16)f0.w;
                t[4]=(bf16)f1.x; t[5]=(bf16)f1.y; t[6]=(bf16)f1.z; t[7]=(bf16)f1.w;
                ra0 = t;
            }
            {
                const float4* p = (const float4*)(Af + (size_t)gmA1 * Kd + gk);
                float4 f0 = p[0], f1 = p[1];
                v8bf t;
                t[0]=(bf16)f0.x; t[1]=(bf16)f0.y; t[2]=(bf16)f0.z; t[3]=(bf16)f0.w;
                t[4]=(bf16)f1.x; t[5]=(bf16)f1.y; t[6]=(bf16)f1.z; t[7]=(bf16)f1.w;
                ra1 = t;
            }
        } else {
            const bf16* Ab = (const bf16*)A;
            ra0 = *(const v8bf*)(Ab + (size_t)gmA0 * Kd + gk);
            ra1 = *(const v8bf*)(Ab + (size_t)gmA1 * Kd + gk);
        }
        rb = *(const v8bf*)(Bt + (size_t)gnB * Kd + gk);
    };
    // stage K-remainder step (guarded, zero-padded)
    auto stage_tail = [&](int step) {
        const int gk = (step << 5) + lk;
        const float* Af = (const float*)A;
        const bf16*  Ab = (const bf16*)A;
        for (int j = 0; j < 8; ++j) {
            const bool ok = (gk + j) < Kd;
            if (AF32) {
                ra0[j] = (bf16)(ok ? Af[(size_t)gmA0*Kd + gk + j] : 0.f);
                ra1[j] = (bf16)(ok ? Af[(size_t)gmA1*Kd + gk + j] : 0.f);
            } else {
                ra0[j] = ok ? Ab[(size_t)gmA0*Kd + gk + j] : (bf16)0.f;
                ra1[j] = ok ? Ab[(size_t)gmA1*Kd + gk + j] : (bf16)0.f;
            }
            rb[j] = ok ? Bt[(size_t)gnB*Kd + gk + j] : (bf16)0.f;
        }
    };
    auto stage = [&](int step) {
        if (step < nfull) stage_fast(step); else stage_tail(step);
    };

    stage(0);
    for (int s = 0; s < nsteps; ++s) {
        __syncthreads();                       // LDS free from previous step
        *(v8bf*)&As[lr][lk]      = ra0;
        *(v8bf*)&As[lr + 64][lk] = ra1;
        *(v8bf*)&Bs[lr][lk]      = rb;
        __syncthreads();
        if (s + 1 < nsteps) stage(s + 1);      // prefetch overlaps WMMAs below

        union { v16bf v; v8bf h[2]; } fa0, fa1, fb0, fb1;
        const int ar = wr*32 + r;
        fa0.h[0] = *(const v8bf*)&As[ar][half*8];
        fa0.h[1] = *(const v8bf*)&As[ar][16 + half*8];
        fa1.h[0] = *(const v8bf*)&As[ar + 16][half*8];
        fa1.h[1] = *(const v8bf*)&As[ar + 16][16 + half*8];
        const int bc = wc*32 + r;
        fb0.h[0] = *(const v8bf*)&Bs[bc][half*8];
        fb0.h[1] = *(const v8bf*)&Bs[bc][16 + half*8];
        fb1.h[0] = *(const v8bf*)&Bs[bc + 16][half*8];
        fb1.h[1] = *(const v8bf*)&Bs[bc + 16][16 + half*8];
        acc00 = __builtin_amdgcn_wmma_f32_16x16x32_bf16(false, fa0.v, false, fb0.v,
                                                        (short)0, acc00, false, false);
        acc01 = __builtin_amdgcn_wmma_f32_16x16x32_bf16(false, fa0.v, false, fb1.v,
                                                        (short)0, acc01, false, false);
        acc10 = __builtin_amdgcn_wmma_f32_16x16x32_bf16(false, fa1.v, false, fb0.v,
                                                        (short)0, acc10, false, false);
        acc11 = __builtin_amdgcn_wmma_f32_16x16x32_bf16(false, fa1.v, false, fb1.v,
                                                        (short)0, acc11, false, false);
    }

    // ---- epilogue: D element i of lane -> row = 8*half + i, col = r ----
    for (int t = 0; t < 2; ++t) {
        const int col = n0 + wc*32 + t*16 + r;
        if (col >= N) continue;
        const float bv = bias ? bias[col] : 0.f;
        for (int s = 0; s < 2; ++s) {
            v8f a = (s == 0) ? (t == 0 ? acc00 : acc01)
                             : (t == 0 ? acc10 : acc11);
            const int rowb = m0 + wr*32 + s*16 + half*8;
            for (int i = 0; i < 8; ++i) {
                const int row = rowb + i;
                if (row < M) {
                    const size_t idx = (size_t)row * N + col;
                    float v = alpha * a[i] + bv;
                    if (accum) v += outF[idx];
                    if (relu)  v = v > 0.f ? v : 0.f;
                    if (outF)  outF[idx]  = v;
                    if (outBf) outBf[idx] = (bf16)v;
                }
            }
        }
    }
}

// ---------------------------------------------------------------------------
// Transpose + convert: dst[C,R] (bf16) = src[R,C]^T (f32)
// ---------------------------------------------------------------------------
__global__ void __launch_bounds__(256)
k_transpose_bf16(const float* __restrict__ src, bf16* __restrict__ dst, int R, int C)
{
    __shared__ float tile[32][33];
    const int c0 = blockIdx.x * 32, r0 = blockIdx.y * 32;
    const int tc = threadIdx.x & 31, tg = threadIdx.x >> 5;
    for (int i = 0; i < 4; ++i) {
        const int rr = tg + i*8;
        const int r_ = r0 + rr;
        tile[rr][tc] = (r_ < R && (c0 + tc) < C) ? src[(size_t)r_ * C + c0 + tc] : 0.f;
    }
    __syncthreads();
    for (int i = 0; i < 4; ++i) {
        const int cc = tg + i*8;
        const int c = c0 + cc, r = r0 + tc;
        if (c < C && r < R) dst[(size_t)c * R + r] = (bf16)tile[tc][cc];
    }
}

// ---------------------------------------------------------------------------
// Small / elementwise kernels
// ---------------------------------------------------------------------------
__global__ void k_fill(float* p, int n, float v)
{ int i = blockIdx.x*256 + threadIdx.x; if (i < n) p[i] = v; }

__global__ void k_reparam(const float* __restrict__ mu, const float* __restrict__ lv,
                          const float* __restrict__ eps, float* __restrict__ z,
                          bf16* __restrict__ zbf, int n)
{
    int i = blockIdx.x*256 + threadIdx.x;
    if (i < n) {
        float v = mu[i] + eps[i] * __expf(0.5f * lv[i]);
        z[i] = v;
        if (zbf) zbf[i] = (bf16)v;
    }
}

// per-row: bf16 mu, bf16 sd = exp(0.5*lv), rn = sum(mu^2)+sum(sd^2)   (Z=64)
__global__ void __launch_bounds__(256)
k_gauss_prep(const float* __restrict__ mu, const float* __restrict__ lv,
             bf16* __restrict__ mubf, bf16* __restrict__ sdbf,
             float* __restrict__ rn, int rows)
{
    int row  = blockIdx.x*8 + (threadIdx.x >> 5);
    int lane = threadIdx.x & 31;
    if (row >= rows) return;
    float s = 0.f;
    for (int j = lane; j < 64; j += 32) {
        float m  = mu[row*64 + j];
        float sd = __expf(0.5f * lv[row*64 + j]);
        mubf[row*64 + j] = (bf16)m;
        sdbf[row*64 + j] = (bf16)sd;
        s += m*m + sd*sd;
    }
    for (int o = 16; o > 0; o >>= 1) s += __shfl_xor(s, o, 32);
    if (!lane) rn[row] = s;
}

__global__ void k_finish_dist(const float* __restrict__ G, const float* __restrict__ rs,
                              const float* __restrict__ rd, float* __restrict__ c,
                              bf16* __restrict__ cbf)
{
    int idx = blockIdx.x*256 + threadIdx.x;   // 1024*1024
    int i = idx >> 10, j = idx & 1023;
    float v = G[idx] + rs[i] + rd[j] + 1e-6f;
    c[idx] = v;
    if (cbf) cbf[idx] = (bf16)v;
}

__global__ void __launch_bounds__(256)
k_rowsumsq(const float* __restrict__ X, float* __restrict__ out, int rows, int cols, float scale)
{
    int row  = blockIdx.x*8 + (threadIdx.x >> 5);
    int lane = threadIdx.x & 31;
    if (row >= rows) return;
    float s = 0.f;
    for (int j = lane; j < cols; j += 32) { float v = X[(size_t)row*cols + j]; s += v*v; }
    for (int o = 16; o > 0; o >>= 1) s += __shfl_xor(s, o, 32);
    if (!lane) out[row] = s * scale;
}

__global__ void k_make_cost(const float* __restrict__ core, const float* __restrict__ f1,
                            const float* __restrict__ f2, const float* __restrict__ cpp,
                            float* __restrict__ cost)
{
    int idx = blockIdx.x*256 + threadIdx.x;
    int i = idx >> 10, j = idx & 1023;
    cost[idx] = 0.1f*(f1[i] + f2[j] - 2.f*core[idx]) + 0.9f*cpp[idx];
}

__global__ void __launch_bounds__(256)
k_maxabs(const float* __restrict__ x, int n, float* out)
{
    __shared__ float sm[256];
    float m = 0.f;
    for (int i = blockIdx.x*256 + threadIdx.x; i < n; i += gridDim.x*256)
        m = fmaxf(m, fabsf(x[i]));
    sm[threadIdx.x] = m; __syncthreads();
    for (int o = 128; o > 0; o >>= 1) {
        if (threadIdx.x < o) sm[threadIdx.x] = fmaxf(sm[threadIdx.x], sm[threadIdx.x + o]);
        __syncthreads();
    }
    if (!threadIdx.x) atomicMax((unsigned int*)out, __float_as_uint(sm[0]));
}

__global__ void k_kernel_exp(const float* __restrict__ cost, const float* __restrict__ mx,
                             const float* __restrict__ tran, float* __restrict__ Km, int n)
{
    int i = blockIdx.x*256 + threadIdx.x;
    if (i < n) Km[i] = __expf(-cost[i] / mx[0]) * tran[i];
}

// out[j] = num / sum_i Km[i,j]*d[i]     (thread per column, coalesced)
__global__ void k_colmatvec(const float* __restrict__ Km, const float* __restrict__ d,
                            float* __restrict__ out, int n, float num)
{
    int j = blockIdx.x*256 + threadIdx.x;
    if (j >= n) return;
    float s = 0.f;
    for (int i = 0; i < n; ++i) s += Km[(size_t)i*n + j] * d[i];
    out[j] = num / s;
}

// out[i] = num / sum_j Km[i,j]*b[j]     (wave per row)
__global__ void __launch_bounds__(256)
k_rowmatvec(const float* __restrict__ Km, const float* __restrict__ bv,
            float* __restrict__ out, int n, float num)
{
    int row  = blockIdx.x*8 + (threadIdx.x >> 5);
    int lane = threadIdx.x & 31;
    if (row >= n) return;
    float s = 0.f;
    for (int j = lane; j < n; j += 32) s += Km[(size_t)row*n + j] * bv[j];
    for (int o = 16; o > 0; o >>= 1) s += __shfl_xor(s, o, 32);
    if (!lane) out[row] = num / s;
}

__global__ void k_tran_update(const float* __restrict__ dual, const float* __restrict__ bv,
                              const float* __restrict__ Km, float* __restrict__ tran)
{
    int idx = blockIdx.x*256 + threadIdx.x;
    int i = idx >> 10, j = idx & 1023;
    tran[idx] = dual[i] * bv[j] * Km[idx];
}

__global__ void k_nan_or(const float* __restrict__ x, int n, int* flag)
{
    int i = blockIdx.x*256 + threadIdx.x;
    if (i < n) { float v = x[i]; if (v != v) atomicOr(flag, 1); }
}

__global__ void k_nan_fix(float* __restrict__ x, int n, const int* __restrict__ flag, float val)
{
    int i = blockIdx.x*256 + threadIdx.x;
    if (i < n && flag[0]) x[i] = val;
}

__global__ void __launch_bounds__(256)
k_align(const float* __restrict__ core, const float* __restrict__ f1,
        const float* __restrict__ f2, const float* __restrict__ cpp,
        const float* __restrict__ tran, float* __restrict__ out)
{
    __shared__ float sm[256];
    float s = 0.f;
    for (int idx = blockIdx.x*256 + threadIdx.x; idx < 1024*1024; idx += gridDim.x*256) {
        int i = idx >> 10, j = idx & 1023;
        float c = 0.1f*(f1[i] + f2[j] - 2.f*core[idx]) + 0.9f*cpp[idx];
        s += c * tran[idx];
    }
    sm[threadIdx.x] = s; __syncthreads();
    for (int o = 128; o > 0; o >>= 1) {
        if (threadIdx.x < o) sm[threadIdx.x] += sm[threadIdx.x + o];
        __syncthreads();
    }
    if (!threadIdx.x) atomicAdd(out, sm[0]);
}

// KL(q||GMM prior), summed over batch; wave per sample, streaming logsumexp over K=20.
__global__ void __launch_bounds__(256)
k_kl_gmm(const float* __restrict__ z, const float* __restrict__ qm,
         const float* __restrict__ qlv, const float* __restrict__ pm,
         const float* __restrict__ plv, float* __restrict__ out)
{
    int b    = blockIdx.x*8 + (threadIdx.x >> 5);
    int lane = threadIdx.x & 31;
    if (b >= 1024) return;
    const float* zb = z + b*64;

    float s = 0.f;
    for (int j = lane; j < 64; j += 32) {
        float lv = qlv[b*64 + j];
        float d  = zb[j] - qm[b*64 + j];
        s += lv + d*d*__expf(-lv);
    }
    for (int o = 16; o > 0; o >>= 1) s += __shfl_xor(s, o, 32);
    float log_q = -0.5f * (64.f * LOG2PI_F + s);

    float m = -1e30f, acc = 0.f;
    for (int k = 0; k < 20; ++k) {
        float t = 0.f;
        for (int j = lane; j < 64; j += 32) {
            float lv = plv[k*64 + j];
            float d  = zb[j] - pm[k*64 + j];
            t += lv + d*d*__expf(-lv);
        }
        for (int o = 16; o > 0; o >>= 1) t += __shfl_xor(t, o, 32);
        float lp = -0.5f * (64.f * LOG2PI_F + t);
        if (lp > m) { acc = acc * __expf(m - lp) + 1.f; m = lp; }
        else        { acc += __expf(lp - m); }
    }
    float log_p = m + logf(acc) - logf(20.f);
    if (!lane) atomicAdd(out, log_q - log_p);
}

__global__ void __launch_bounds__(256)
k_sumsq_diff(const float* __restrict__ a, const float* __restrict__ b, int n, float* accum)
{
    __shared__ float sm[256];
    float s = 0.f;
    for (int i = blockIdx.x*256 + threadIdx.x; i < n; i += gridDim.x*256) {
        float d = a[i] - b[i]; s += d*d;
    }
    sm[threadIdx.x] = s; __syncthreads();
    for (int o = 128; o > 0; o >>= 1) {
        if (threadIdx.x < o) sm[threadIdx.x] += sm[threadIdx.x + o];
        __syncthreads();
    }
    if (!threadIdx.x) atomicAdd(accum, sm[0]);
}

__global__ void k_sqrt_to(float* dst, const float* src)
{ if (threadIdx.x == 0) dst[0] = sqrtf(src[0]); }

// ---------------------------------------------------------------------------
// Host orchestration
// ---------------------------------------------------------------------------
extern "C" void kernel_launch(void* const* d_in, const int* in_sizes, int n_in,
                              void* d_out, int out_size, void* d_ws, size_t ws_size,
                              hipStream_t stream)
{
    (void)in_sizes; (void)n_in; (void)out_size; (void)ws_size;
    const int B = 1024, NM = 20000, NB = 25000, Z = 64, H = 256;

    const float* x      = (const float*)d_in[2];
    const float* y      = (const float*)d_in[3];
    const float* p_mu_a = (const float*)d_in[4];
    const float* p_lv_a = (const float*)d_in[5];
    const float* p_mu_b = (const float*)d_in[6];
    const float* p_lv_b = (const float*)d_in[7];
    const float* eps_x  = (const float*)d_in[8];
    const float* eps_y  = (const float*)d_in[9];
    const float* eps_xr = (const float*)d_in[10];
    const float* eps_yr = (const float*)d_in[11];
    const float *W1a=(const float*)d_in[12], *b1a=(const float*)d_in[13];
    const float *W21a=(const float*)d_in[14], *b21a=(const float*)d_in[15];
    const float *W22a=(const float*)d_in[16], *b22a=(const float*)d_in[17];
    const float *W3a=(const float*)d_in[18], *b3a=(const float*)d_in[19];
    const float *W4a=(const float*)d_in[20], *b4a=(const float*)d_in[21];
    const float *W1b=(const float*)d_in[22], *b1b=(const float*)d_in[23];
    const float *W21b=(const float*)d_in[24], *b21b=(const float*)d_in[25];
    const float *W22b=(const float*)d_in[26], *b22b=(const float*)d_in[27];
    const float *W3b=(const float*)d_in[28], *b3b=(const float*)d_in[29];
    const float *W4b=(const float*)d_in[30], *b4b=(const float*)d_in[31];

    float* outp = (float*)d_out;
    const size_t O_PX = 0;
    const size_t O_PY = O_PX + (size_t)B*NM;
    const size_t O_PX2Y = O_PY + (size_t)B*NB;
    const size_t O_PY2X = O_PX2Y + (size_t)B*NB;
    const size_t O_ZX = O_PY2X + (size_t)B*NM;
    const size_t O_ZY = O_ZX + (size_t)B*Z;
    const size_t O_GA = O_ZY + (size_t)B*Z;
    const size_t O_GB = O_GA + 1;
    const size_t O_FXR = O_GB + 1;
    const size_t O_FYR = O_FXR + (size_t)B*Z;
    const size_t O_AL  = O_FYR + (size_t)B*Z;
    const size_t O_CX  = O_AL + 1;
    const size_t O_CY  = O_CX + 1;

    // ---- workspace bump allocator (deterministic) ----
    char* base = (char*)d_ws;
    size_t cur = 0;
    auto aBF = [&](size_t n)->bf16* { bf16* p=(bf16*)(base+cur); cur += ((n*2+255)/256)*256; return p; };
    auto aF  = [&](size_t n)->float*{ float* p=(float*)(base+cur); cur += ((n*4+255)/256)*256; return p; };

    bf16 *W1aT=aBF((size_t)H*NM), *W21aT=aBF(Z*H), *W22aT=aBF(Z*H), *W3aT=aBF(H*Z), *W4aT=aBF((size_t)NM*H);
    bf16 *W1bT=aBF((size_t)H*NB), *W21bT=aBF(Z*H), *W22bT=aBF(Z*H), *W3bT=aBF(H*Z), *W4bT=aBF((size_t)NB*H);
    bf16 *hA=aBF(B*H), *hB=aBF(B*H), *hdA=aBF(B*H), *hdB=aBF(B*H);
    bf16 *hdX2Y=aBF(B*H), *hdY2X=aBF(B*H), *hXR=aBF(B*H), *hYR=aBF(B*H);
    float *mu_a=aF(B*Z), *lv_a=aF(B*Z), *mu_b=aF(B*Z), *lv_b=aF(B*Z);
    float *m_xr=aF(B*Z), *lv_xr=aF(B*Z), *m_yr=aF(B*Z), *lv_yr=aF(B*Z);
    bf16 *zx_bf=aBF(B*Z), *zy_bf=aBF(B*Z);
    bf16 *px2y_bf=aBF((size_t)B*NB), *py2x_bf=aBF((size_t)B*NM);
    bf16 *muS=aBF(B*Z), *sdS=aBF(B*Z), *muD=aBF(B*Z), *sdD=aBF(B*Z);
    float *rn_s=aF(1024), *rn_d=aF(1024), *f1=aF(1024), *f2=aF(1024);
    const size_t SQ = (size_t)1024*1024;
    float *G=aF(SQ), *c_post=aF(SQ), *c_prior=aF(SQ), *c_pp=aF(SQ);
    bf16  *cpost_bf=aBF(SQ), *cprior_bf=aBF(SQ), *T2t_bf=aBF(SQ);
    float *tran=aF(SQ), *T2=aF(SQ), *core=aF(SQ), *costm=aF(SQ), *Km=aF(SQ);
    float *dual=aF(1024), *bvec=aF(1024), *scal=aF(8);   // 0:maxabs 1:ssx 2:ssy 3:nanflag

    auto gemm = [&](const void* A, int aF32, const bf16* Bt, const float* bias,
                    float* oF, bf16* oB, float alpha, int relu, int accf,
                    int M, int N, int Kd) {
        dim3 g((N + 63)/64, (M + 127)/128);
        if (aF32)
            k_gemm_t<1><<<g, 256, 0, stream>>>(A, Bt, bias, oF, oB, alpha, relu, accf, M, N, Kd);
        else
            k_gemm_t<0><<<g, 256, 0, stream>>>(A, Bt, bias, oF, oB, alpha, relu, accf, M, N, Kd);
    };
    auto tr = [&](const float* src, bf16* dst, int R, int C) {
        dim3 g((C + 31)/32, (R + 31)/32);
        k_transpose_bf16<<<g, 256, 0, stream>>>(src, dst, R, C);
    };
    const int EW = (1024*1024)/256;   // blocks for 1024^2 elementwise

    // zero accumulators / scalar outputs
    k_fill<<<1,32,0,stream>>>(scal, 8, 0.f);
    k_fill<<<1,32,0,stream>>>(outp + O_GA, 2, 0.f);
    k_fill<<<1,32,0,stream>>>(outp + O_AL, 1, 0.f);

    // ---- weight transposes (f32 -> bf16, K-major) ----
    tr(W1a, W1aT, NM, H);  tr(W21a, W21aT, H, Z); tr(W22a, W22aT, H, Z);
    tr(W3a, W3aT, Z, H);   tr(W4a, W4aT, H, NM);
    tr(W1b, W1bT, NB, H);  tr(W21b, W21bT, H, Z); tr(W22b, W22bT, H, Z);
    tr(W3b, W3bT, Z, H);   tr(W4b, W4bT, H, NB);

    // ---- encoders ----
    gemm(x, 1, W1aT, b1a, nullptr, hA, 1.f, 1, 0, B, H, NM);
    gemm(hA, 0, W21aT, b21a, mu_a, nullptr, 1.f, 0, 0, B, Z, H);
    gemm(hA, 0, W22aT, b22a, lv_a, nullptr, 1.f, 0, 0, B, Z, H);
    k_reparam<<<(B*Z+255)/256,256,0,stream>>>(mu_a, lv_a, eps_x, outp + O_ZX, zx_bf, B*Z);

    gemm(y, 1, W1bT, b1b, nullptr, hB, 1.f, 1, 0, B, H, NB);
    gemm(hB, 0, W21bT, b21b, mu_b, nullptr, 1.f, 0, 0, B, Z, H);
    gemm(hB, 0, W22bT, b22b, lv_b, nullptr, 1.f, 0, 0, B, Z, H);
    k_reparam<<<(B*Z+255)/256,256,0,stream>>>(mu_b, lv_b, eps_y, outp + O_ZY, zy_bf, B*Z);

    // ---- decoders (same + cross) ----
    gemm(zx_bf, 0, W3aT, b3a, nullptr, hdA, 1.f, 1, 0, B, H, Z);
    gemm(hdA, 0, W4aT, b4a, outp + O_PX, nullptr, 1.f, 0, 0, B, NM, H);
    gemm(zy_bf, 0, W3bT, b3b, nullptr, hdB, 1.f, 1, 0, B, H, Z);
    gemm(hdB, 0, W4bT, b4b, outp + O_PY, nullptr, 1.f, 0, 0, B, NB, H);
    gemm(zx_bf, 0, W3bT, b3b, nullptr, hdX2Y, 1.f, 1, 0, B, H, Z);
    gemm(hdX2Y, 0, W4bT, b4b, outp + O_PX2Y, px2y_bf, 1.f, 0, 0, B, NB, H);
    gemm(zy_bf, 0, W3aT, b3a, nullptr, hdY2X, 1.f, 1, 0, B, H, Z);
    gemm(hdY2X, 0, W4aT, b4a, outp + O_PY2X, py2x_bf, 1.f, 0, 0, B, NM, H);

    // ---- cycle re-encode ----
    gemm(px2y_bf, 0, W1bT, b1b, nullptr, hXR, 1.f, 1, 0, B, H, NB);
    gemm(hXR, 0, W21bT, b21b, m_xr, nullptr, 1.f, 0, 0, B, Z, H);
    gemm(hXR, 0, W22bT, b22b, lv_xr, nullptr, 1.f, 0, 0, B, Z, H);
    k_reparam<<<(B*Z+255)/256,256,0,stream>>>(m_xr, lv_xr, eps_xr, outp + O_FXR, nullptr, B*Z);
    gemm(py2x_bf, 0, W1aT, b1a, nullptr, hYR, 1.f, 1, 0, B, H, NM);
    gemm(hYR, 0, W21aT, b21a, m_yr, nullptr, 1.f, 0, 0, B, Z, H);
    gemm(hYR, 0, W22aT, b22a, lv_yr, nullptr, 1.f, 0, 0, B, Z, H);
    k_reparam<<<(B*Z+255)/256,256,0,stream>>>(m_yr, lv_yr, eps_yr, outp + O_FYR, nullptr, B*Z);

    // ---- cycle losses ----
    k_sumsq_diff<<<64,256,0,stream>>>(outp + O_ZX, outp + O_FXR, B*Z, scal + 1);
    k_sqrt_to<<<1,32,0,stream>>>(outp + O_CX, scal + 1);
    k_sumsq_diff<<<64,256,0,stream>>>(outp + O_ZY, outp + O_FYR, B*Z, scal + 2);
    k_sqrt_to<<<1,32,0,stream>>>(outp + O_CY, scal + 2);

    // ---- GMM KL losses ----
    k_kl_gmm<<<128,256,0,stream>>>(outp + O_ZX, mu_a, lv_a, p_mu_a, p_lv_a, outp + O_GA);
    k_kl_gmm<<<128,256,0,stream>>>(outp + O_ZY, mu_b, lv_b, p_mu_b, p_lv_b, outp + O_GB);

    // ==================== FGW: _fgw(mu_b, mu_a, lv_b, lv_a) ====================
    k_gauss_prep<<<128,256,0,stream>>>(mu_b, lv_b, muS, sdS, rn_s, 1024);
    k_gauss_prep<<<128,256,0,stream>>>(mu_a, lv_a, muD, sdD, rn_d, 1024);

    // c_post = dist(S,S), c_prior = dist(D,D), c_pp = dist(S,D)
    gemm(muS, 0, muS, nullptr, G, nullptr, -2.f, 0, 0, 1024, 1024, 64);
    gemm(sdS, 0, sdS, nullptr, G, nullptr, -2.f, 0, 1, 1024, 1024, 64);
    k_finish_dist<<<EW,256,0,stream>>>(G, rn_s, rn_s, c_post, cpost_bf);
    gemm(muD, 0, muD, nullptr, G, nullptr, -2.f, 0, 0, 1024, 1024, 64);
    gemm(sdD, 0, sdD, nullptr, G, nullptr, -2.f, 0, 1, 1024, 1024, 64);
    k_finish_dist<<<EW,256,0,stream>>>(G, rn_d, rn_d, c_prior, cprior_bf);
    gemm(muS, 0, muD, nullptr, G, nullptr, -2.f, 0, 0, 1024, 1024, 64);
    gemm(sdS, 0, sdD, nullptr, G, nullptr, -2.f, 0, 1, 1024, 1024, 64);
    k_finish_dist<<<EW,256,0,stream>>>(G, rn_s, rn_d, c_pp, nullptr);

    k_rowsumsq<<<128,256,0,stream>>>(c_post, f1, 1024, 1024, 1.f/1024.f);
    k_rowsumsq<<<128,256,0,stream>>>(c_prior, f2, 1024, 1024, 1.f/1024.f);

    k_fill<<<EW,256,0,stream>>>(tran, 1024*1024, 1.f/(1024.f*1024.f));
    k_fill<<<4,256,0,stream>>>(dual, 1024, 1.f/1024.f);

    for (int it = 0; it < 10; ++it) {
        // core = c_post @ (tran @ c_prior^T)
        gemm(tran, 1, cprior_bf, nullptr, T2, nullptr, 1.f, 0, 0, 1024, 1024, 1024);
        tr(T2, T2t_bf, 1024, 1024);
        gemm(cpost_bf, 0, T2t_bf, nullptr, core, nullptr, 1.f, 0, 0, 1024, 1024, 1024);
        k_make_cost<<<EW,256,0,stream>>>(core, f1, f2, c_pp, costm);
        k_fill<<<1,32,0,stream>>>(scal, 1, 0.f);
        k_maxabs<<<256,256,0,stream>>>(costm, 1024*1024, scal);
        k_kernel_exp<<<EW,256,0,stream>>>(costm, scal, tran, Km, 1024*1024);
        k_colmatvec<<<4,256,0,stream>>>(Km, dual, bvec, 1024, 1.f/1024.f);
        for (int in5 = 0; in5 < 5; ++in5) {
            k_rowmatvec<<<128,256,0,stream>>>(Km, bvec, dual, 1024, 1.f/1024.f);
            k_colmatvec<<<4,256,0,stream>>>(Km, dual, bvec, 1024, 1.f/1024.f);
        }
        k_tran_update<<<EW,256,0,stream>>>(dual, bvec, Km, tran);
    }

    // NaN guard on tran
    k_fill<<<1,32,0,stream>>>(scal + 3, 1, 0.f);
    k_nan_or<<<EW,256,0,stream>>>(tran, 1024*1024, (int*)(scal + 3));
    k_nan_fix<<<EW,256,0,stream>>>(tran, 1024*1024, (const int*)(scal + 3),
                                   1.f/(1024.f*1024.f));

    // final cost with detached tran; align_global_loss = sum(cost*tran)
    gemm(tran, 1, cprior_bf, nullptr, T2, nullptr, 1.f, 0, 0, 1024, 1024, 1024);
    tr(T2, T2t_bf, 1024, 1024);
    gemm(cpost_bf, 0, T2t_bf, nullptr, core, nullptr, 1.f, 0, 0, 1024, 1024, 1024);
    k_align<<<512,256,0,stream>>>(core, f1, f2, c_pp, tran, outp + O_AL);
}
